// DeformConv_1340029796639
// MI455X (gfx1250) — matrix operand order repaired and checked by reading, not compiled, not run on hardware
//
#include <hip/hip_runtime.h>
#include <hip/hip_bf16.h>
#include <math.h>

// ---------------------------------------------------------------------------
// Deformable Conv v2 (modulated) + BN + ReLU for MI455X (gfx1250, wave32).
// Main contraction runs on v_wmma_f32_16x16x32_bf16 (f32 accumulate).
// ---------------------------------------------------------------------------

#define B_    4
#define CI_   64
#define CO_   64
#define H_    128
#define W_    128
#define K_    9
#define HW_   (H_ * W_)          // 16384
#define CK_   (CI_ * K_)         // 576  (GEMM K dim)
#define SROW  592                // padded LDS row stride (bf16 elems), 16B-aligned rows

typedef __attribute__((ext_vector_type(16))) __bf16 bf16x16;
typedef __attribute__((ext_vector_type(8)))  float  f32x8;

static __device__ __forceinline__ unsigned short f2bf(float f) {
    unsigned int u = __float_as_uint(f);
    unsigned int r = (u + 0x7FFFu + ((u >> 16) & 1u)) >> 16;   // round-nearest-even
    return (unsigned short)r;
}

// ---------------------------------------------------------------------------
// Kernel 0: pack main conv weight (O,C,3,3) fp32 -> bf16 [oc][k*64 + c]
// ---------------------------------------------------------------------------
__global__ void pack_weight_kernel(const float* __restrict__ w,
                                   unsigned short* __restrict__ wbf) {
    int idx = blockIdx.x * blockDim.x + threadIdx.x;       // 0 .. CO_*CK_-1
    if (idx >= CO_ * CK_) return;
    int oc = idx / CK_;
    int r  = idx - oc * CK_;
    int k  = r >> 6;                                       // r / 64
    int c  = r & 63;
    wbf[idx] = f2bf(w[(oc * CI_ + c) * K_ + k]);
}

// ---------------------------------------------------------------------------
// Kernel 1: offset conv (27 out ch, 3x3, pad 1) -> dy, dx, sigmoid(mask)
// One thread per output pixel, all 27 channels. w_off cached in LDS.
// ---------------------------------------------------------------------------
__global__ __launch_bounds__(256)
void offset_conv_kernel(const float* __restrict__ x,
                        const float* __restrict__ w_off,
                        const float* __restrict__ b_off,
                        float* __restrict__ dy,
                        float* __restrict__ dx,
                        float* __restrict__ mk) {
    __shared__ float ws[27 * CI_ * K_];                    // 15552 floats = 62 KB
    for (int i = threadIdx.x; i < 27 * CI_ * K_; i += 256)
        ws[i] = w_off[i];
    __syncthreads();

    int idx = blockIdx.x * 256 + threadIdx.x;              // B*H*W threads
    int b   = idx >> 14;
    int rem = idx & (HW_ - 1);
    int ho  = rem >> 7;
    int wo  = rem & (W_ - 1);

    float acc[27];
#pragma unroll
    for (int oc = 0; oc < 27; ++oc) acc[oc] = 0.0f;

    const float* xb = x + (size_t)b * CI_ * HW_;
#pragma unroll
    for (int ky = 0; ky < 3; ++ky) {
        int y = ho + ky - 1;
        if (y < 0 || y >= H_) continue;
#pragma unroll
        for (int kx = 0; kx < 3; ++kx) {
            int xc = wo + kx - 1;
            if (xc < 0 || xc >= W_) continue;
            int kbase = ky * 3 + kx;
            const float* xp = xb + y * W_ + xc;
            for (int c = 0; c < CI_; ++c) {
                float v = xp[c * HW_];
                const float* wr = &ws[c * K_ + kbase];
#pragma unroll
                for (int oc = 0; oc < 27; ++oc)
                    acc[oc] += v * wr[oc * CI_ * K_];
            }
        }
    }
#pragma unroll
    for (int oc = 0; oc < 27; ++oc) acc[oc] += b_off[oc];

    int pos = (b * K_) * HW_ + ho * W_ + wo;               // [b][k][h][w]
#pragma unroll
    for (int k = 0; k < K_; ++k) {
        dy[pos + k * HW_] = acc[2 * k];
        dx[pos + k * HW_] = acc[2 * k + 1];
        float m = acc[18 + k];
        mk[pos + k * HW_] = 1.0f / (1.0f + expf(-m));
    }
}

// ---------------------------------------------------------------------------
// Kernel 2: deformable gather + WMMA GEMM.
// Block = 128 threads (4 waves) handles (b, ho, 16 pixels) -> all 64 out ch.
// ---------------------------------------------------------------------------
__global__ __launch_bounds__(128)
void deform_gemm_kernel(const float* __restrict__ x,
                        const unsigned short* __restrict__ wbf,
                        const float* __restrict__ bias,
                        const float* __restrict__ dy,
                        const float* __restrict__ dx,
                        const float* __restrict__ mk,
                        float* __restrict__ out) {
    __shared__ int   sO[4][K_ * 16];                       // corner flat offsets
    __shared__ float sW[4][K_ * 16];                       // corner weights * mask
    __shared__ __align__(16) unsigned short Sm[16 * SROW]; // sampled tile, bf16

    int bid = blockIdx.x;                                  // B * H * (W/16)
    int b   = bid >> 10;
    int rem = bid & 1023;
    int ho  = rem >> 3;
    int wo0 = (rem & 7) << 4;
    int tid = threadIdx.x;

    // ---- phase 1: per (k, pixel) bilinear params (mask folded into weights)
    for (int i = tid; i < K_ * 16; i += 128) {
        int k = i >> 4, p = i & 15;
        int wo = wo0 + p;
        int gi = (b * K_ + k) * HW_ + ho * W_ + wo;
        float py = (float)(ho - 1 + (k / 3)) + dy[gi];
        float px = (float)(wo - 1 + (k % 3)) + dx[gi];
        float m  = mk[gi];
        float y0f = floorf(py), x0f = floorf(px);
        float wy1 = py - y0f,  wx1 = px - x0f;
        float wy0 = 1.0f - wy1, wx0 = 1.0f - wx1;
        int iy0 = (int)y0f, ix0 = (int)x0f;
        int iy1 = iy0 + 1,  ix1 = ix0 + 1;
        float vy0 = (iy0 >= 0 && iy0 < H_) ? 1.0f : 0.0f;
        float vy1 = (iy1 >= 0 && iy1 < H_) ? 1.0f : 0.0f;
        float vx0 = (ix0 >= 0 && ix0 < W_) ? 1.0f : 0.0f;
        float vx1 = (ix1 >= 0 && ix1 < W_) ? 1.0f : 0.0f;
        int cy0 = min(max(iy0, 0), H_ - 1), cy1 = min(max(iy1, 0), H_ - 1);
        int cx0 = min(max(ix0, 0), W_ - 1), cx1 = min(max(ix1, 0), W_ - 1);
        sO[0][i] = cy0 * W_ + cx0;  sW[0][i] = wy0 * wx0 * m * vy0 * vx0;
        sO[1][i] = cy0 * W_ + cx1;  sW[1][i] = wy0 * wx1 * m * vy0 * vx1;
        sO[2][i] = cy1 * W_ + cx0;  sW[2][i] = wy1 * wx0 * m * vy1 * vx0;
        sO[3][i] = cy1 * W_ + cx1;  sW[3][i] = wy1 * wx1 * m * vy1 * vx1;
    }
    __syncthreads();

    // ---- phase 2: gather sampled tile -> LDS bf16, layout Sm[p][k*64 + c]
    {
        int p  = tid & 15;
        int cg = tid >> 4;                                 // 0..7
        const float* xb = x + (size_t)b * CI_ * HW_;
#pragma unroll
        for (int k = 0; k < K_; ++k) {
            int i = (k << 4) + p;
            int o00 = sO[0][i], o01 = sO[1][i], o10 = sO[2][i], o11 = sO[3][i];
            float w00 = sW[0][i], w01 = sW[1][i], w10 = sW[2][i], w11 = sW[3][i];
            unsigned short* srow = &Sm[p * SROW + (k << 6)];
#pragma unroll
            for (int j = 0; j < 8; ++j) {
                int c = (cg << 3) + j;
                const float* xc = xb + c * HW_;
                float s = xc[o00] * w00 + xc[o01] * w01 +
                          xc[o10] * w10 + xc[o11] * w11;
                srow[c] = f2bf(s);
            }
        }
    }
    __syncthreads();

    // ---- phase 3: WMMA GEMM  D[16oc x 16px] per wave, K = 576 in 18 steps
    int wv   = tid >> 5;                                   // wave id: M-tile
    int lane = tid & 31;
    int half = lane >> 4;
    int nm   = lane & 15;                                  // A: M row / B: N col

    const unsigned short* arow = wbf + (wv * 16 + nm) * CK_;
    const unsigned short* brow = &Sm[nm * SROW];

    f32x8 acc = {0.f, 0.f, 0.f, 0.f, 0.f, 0.f, 0.f, 0.f};
    union Vec { uint4 q[2]; bf16x16 v; };
#pragma unroll
    for (int kt = 0; kt < CK_ / 32; ++kt) {
        Vec A, Bm;
        int ka = kt * 32 + half * 8;                       // A: K = ka+e / ka+16+e
        A.q[0] = *(const uint4*)(arow + ka);
        A.q[1] = *(const uint4*)(arow + ka + 16);
        int kb = kt * 32 + half * 8;                       // B: K = kb+e / kb+16+e
        Bm.q[0] = *(const uint4*)(brow + kb);
        Bm.q[1] = *(const uint4*)(brow + kb + 16);
        acc = __builtin_amdgcn_wmma_f32_16x16x32_bf16(
                  false, A.v, false, Bm.v, (short)0, acc, false, false);
    }

    // ---- phase 4: bias + store (D layout: VGPR v -> M = v + 8*half)
#pragma unroll
    for (int v = 0; v < 8; ++v) {
        int oc = wv * 16 + half * 8 + v;
        float val = acc[v] + bias[oc];
        out[((size_t)(b * CO_ + oc) << 14) + (ho << 7) + wo0 + nm] = val;
    }
}

// ---------------------------------------------------------------------------
// Kernel 3: per-channel BN statistics (mean, var) over B*H*W
// ---------------------------------------------------------------------------
__global__ __launch_bounds__(256)
void bn_stats_kernel(const float* __restrict__ cout, float* __restrict__ mv) {
    __shared__ float rs[256], rq[256];
    int oc = blockIdx.x;
    float s = 0.f, q = 0.f;
    for (int i = threadIdx.x; i < B_ * HW_; i += 256) {
        int b = i >> 14, r = i & (HW_ - 1);
        float v = cout[((size_t)(b * CO_ + oc) << 14) + r];
        s += v; q += v * v;
    }
    rs[threadIdx.x] = s; rq[threadIdx.x] = q;
    __syncthreads();
    for (int off = 128; off > 0; off >>= 1) {
        if (threadIdx.x < off) {
            rs[threadIdx.x] += rs[threadIdx.x + off];
            rq[threadIdx.x] += rq[threadIdx.x + off];
        }
        __syncthreads();
    }
    if (threadIdx.x == 0) {
        float inv = 1.0f / (float)(B_ * HW_);
        float mean = rs[0] * inv;
        float var  = rq[0] * inv - mean * mean;
        mv[oc * 2]     = mean;
        mv[oc * 2 + 1] = var;
    }
}

// ---------------------------------------------------------------------------
// Kernel 4: BN apply + ReLU
// ---------------------------------------------------------------------------
__global__ __launch_bounds__(256)
void bn_relu_kernel(const float* __restrict__ cout,
                    const float* __restrict__ mv,
                    const float* __restrict__ gamma,
                    const float* __restrict__ beta,
                    float* __restrict__ out) {
    int idx = blockIdx.x * 256 + threadIdx.x;              // B*CO*H*W
    int oc  = (idx >> 14) & (CO_ - 1);
    float mean = mv[oc * 2], var = mv[oc * 2 + 1];
    float y = (cout[idx] - mean) * rsqrtf(var + 1e-5f) * gamma[oc] + beta[oc];
    out[idx] = fmaxf(y, 0.0f);
}

// ---------------------------------------------------------------------------
extern "C" void kernel_launch(void* const* d_in, const int* in_sizes, int n_in,
                              void* d_out, int out_size, void* d_ws, size_t ws_size,
                              hipStream_t stream) {
    const float* x      = (const float*)d_in[0];
    const float* weight = (const float*)d_in[1];
    const float* bias   = (const float*)d_in[2];
    const float* w_off  = (const float*)d_in[3];
    const float* b_off  = (const float*)d_in[4];
    const float* gamma  = (const float*)d_in[5];
    const float* beta   = (const float*)d_in[6];
    float* out = (float*)d_out;

    char* ws = (char*)d_ws;
    const size_t nOff = (size_t)B_ * K_ * HW_;             // 589824
    float* dy   = (float*)(ws);
    float* dx   = (float*)(ws + nOff * 4);
    float* mk   = (float*)(ws + nOff * 8);
    float* cout = (float*)(ws + nOff * 12);                // B*CO*H*W floats
    unsigned short* wbf = (unsigned short*)(ws + nOff * 12 + (size_t)B_ * CO_ * HW_ * 4);
    float* mv   = (float*)((char*)wbf + (size_t)CO_ * CK_ * 2 + 64);

    // 0) pack main weights to bf16
    pack_weight_kernel<<<(CO_ * CK_ + 255) / 256, 256, 0, stream>>>(weight, wbf);
    // 1) offset conv -> dy, dx, sigmoid(mask)
    offset_conv_kernel<<<(B_ * HW_) / 256, 256, 0, stream>>>(x, w_off, b_off, dy, dx, mk);
    // 2) deformable gather + WMMA GEMM
    deform_gemm_kernel<<<B_ * H_ * (W_ / 16), 128, 0, stream>>>(x, wbf, bias, dy, dx, mk, cout);
    // 3) BN stats
    bn_stats_kernel<<<CO_, 256, 0, stream>>>(cout, mv);
    // 4) BN apply + ReLU
    bn_relu_kernel<<<((size_t)B_ * CO_ * HW_) / 256, 256, 0, stream>>>(cout, mv, gamma, beta, out);
}
